// LSTMModel_24352464570004
// MI455X (gfx1250) — matrix operand order repaired and checked by reading, not compiled, not run on hardware
//
#include <hip/hip_runtime.h>

typedef __attribute__((ext_vector_type(16))) _Float16     v16h;
typedef __attribute__((ext_vector_type(8)))  _Float16     v8h;
typedef __attribute__((ext_vector_type(2)))  _Float16     v2h;
typedef __attribute__((ext_vector_type(8)))  float        v8f;
typedef __attribute__((ext_vector_type(8)))  unsigned int v8u;

#define B_SZ   128
#define T_SZ   1024
#define IN_SZ  7
#define H_SZ   64
#define G4     256   // 4*H
#define KCAT   96    // 64 (h) + 7 (x) + 1 (bias) + 24 pad -> 3 K-chunks of 32
#define OUT_SZ 500
#define OUT_PAD 512

// Packed fragment buffers: [frag][lane(32)][word(8)] u32, one u32 = (f16 k, f16 k+1)
#define NWCAT_W (3 * 4 * 4 * 32 * 8)   // kc(3) x gate(4) x wid(4) x lane x word = 12288 u32
#define NWFC_W  (2 * 8 * 4 * 32 * 8)   // kc(2) x nt8(8) x wid(4) x lane x word = 16384 u32

// workspace layout (bytes)
#define WS_WCAT 0
#define WS_WFC  (NWCAT_W * 4)                   // 49152
#define WS_HBUF (WS_WFC + NWFC_W * 4)           // 114688; Hbuf = f16 [B][T][64]

#if __has_builtin(__builtin_amdgcn_tanhf)
__device__ __forceinline__ float fast_tanh(float v) { return __builtin_amdgcn_tanhf(v); }
#else
__device__ __forceinline__ float fast_tanh(float v) {
    return 1.0f - 2.0f / (__expf(2.0f * v) + 1.0f);
}
#endif
__device__ __forceinline__ float fast_sigmoid(float v) {
    return 0.5f * fast_tanh(0.5f * v) + 0.5f;
}

__device__ __forceinline__ unsigned int pack_h2(float lo, float hi) {
    v2h p;
    p[0] = (_Float16)lo;
    p[1] = (_Float16)hi;
    return __builtin_bit_cast(unsigned int, p);
}

// concatenated recurrence weight element: K = [W_hh^T(64) | W_ih^T(7) | bias(1) | 0-pad]
__device__ __forceinline__ float wcat_elem(int k, int col,
                                           const float* W_hh, const float* W_ih,
                                           const float* b_ih, const float* b_hh) {
    if (k < H_SZ)               return W_hh[col * H_SZ + k];
    if (k < H_SZ + IN_SZ)       return W_ih[col * IN_SZ + (k - H_SZ)];
    if (k == H_SZ + IN_SZ)      return b_ih[col] + b_hh[col];
    return 0.0f;
}

// ---- prep: Wcat in per-lane packed WMMA B-fragment layout ----
__global__ void prep_wcat_kernel(const float* __restrict__ W_hh, const float* __restrict__ W_ih,
                                 const float* __restrict__ b_ih, const float* __restrict__ b_hh,
                                 unsigned int* __restrict__ WcatP) {
    int idx = blockIdx.x * 256 + threadIdx.x;
    if (idx >= NWCAT_W) return;
    const int w    = idx % 8;
    const int lane = (idx / 8) % 32;
    const int wid  = (idx / 256) % 4;
    const int g    = (idx / 1024) % 4;
    const int kc   = idx / 4096;
    const int col  = g * 64 + wid * 16 + (lane & 15);
    const int k0   = kc * 32 + ((lane < 16) ? 0 : 8) + ((w < 4) ? 2 * w : 16 + 2 * (w - 4));
    WcatP[idx] = pack_h2(wcat_elem(k0, col, W_hh, W_ih, b_ih, b_hh),
                         wcat_elem(k0 + 1, col, W_hh, W_ih, b_ih, b_hh));
}

// ---- prep: W_fc^T in per-lane packed WMMA B-fragment layout ----
__global__ void prep_wfc_kernel(const float* __restrict__ W_fc, unsigned int* __restrict__ WfcP) {
    int idx = blockIdx.x * 256 + threadIdx.x;
    if (idx >= NWFC_W) return;
    const int w    = idx % 8;
    const int lane = (idx / 8) % 32;
    const int wid  = (idx / 256) % 4;
    const int nt8  = (idx / 1024) % 8;
    const int kc   = idx / 8192;
    const int col  = (wid * 8 + nt8) * 16 + (lane & 15);
    const int k0   = kc * 32 + ((lane < 16) ? 0 : 8) + ((w < 4) ? 2 * w : 16 + 2 * (w - 4));
    const float lo = (col < OUT_SZ) ? W_fc[col * H_SZ + k0] : 0.0f;
    const float hi = (col < OUT_SZ) ? W_fc[col * H_SZ + k0 + 1] : 0.0f;
    WfcP[idx] = pack_h2(lo, hi);
}

// ---- LSTM recurrence: 8 WGs x 128 threads, 16 batch rows per WG ----
// Wave w owns columns [w*16, w*16+16) of ALL four gates (N-tiles w, w+4, w+8, w+12),
// so the pointwise LSTM update runs entirely in registers. A matrix [h|x|1] is
// double-buffered in LDS -> exactly one barrier per timestep.
__global__ __launch_bounds__(128) void lstm_kernel(const float* __restrict__ x,
                                                   const unsigned int* __restrict__ WcatP,
                                                   _Float16* __restrict__ Hbuf) {
    __shared__ _Float16 A_lds[2][16][KCAT];   // [buf][row][ h(64) | x(7) | 1 | pad ]

    const int tid  = threadIdx.x;
    const int lane = tid & 31;               // wave32
    const int wid  = tid >> 5;               // 4 waves
    const int nlo  = lane & 15;
    const int b0   = blockIdx.x * 16;

    // init both buffers: h=0, bias col=1, pad=0 (x cols rewritten per step)
    for (int idx = tid; idx < 2 * 16 * KCAT; idx += 128) {
        int c = idx % KCAT;
        (&A_lds[0][0][0])[idx] = (c == 71) ? (_Float16)1.0f : (_Float16)0.0f;
    }

    // register-resident B fragments, loaded pre-packed (contiguous 32B per fragment)
    v16h bfrag[3][4];
    {
        const v8u* Wv = (const v8u*)WcatP;
#pragma unroll
        for (int kc = 0; kc < 3; ++kc)
#pragma unroll
            for (int g = 0; g < 4; ++g)
                bfrag[kc][g] = __builtin_bit_cast(
                    v16h, Wv[((kc * 4 + g) * 4 + wid) * 32 + lane]);
    }

    // cell state in registers: lane covers rows rbase..rbase+7, column hcol
    const int rbase = (lane < 16) ? 0 : 8;
    const int hcol  = wid * 16 + nlo;
    float c_st[8];
#pragma unroll
    for (int r = 0; r < 8; ++r) c_st[r] = 0.0f;

    // coalesced Hbuf store mapping: thread -> (row srow, cols scb..scb+7)
    const int srow = tid >> 3;
    const int scb  = (tid & 7) * 8;
    _Float16* const hrow_out = Hbuf + (size_t)(b0 + srow) * T_SZ * H_SZ + scb;

    // x(0) into buffer 0 (threads 0..111 cover 16 rows x 7 features)
    const int  xr = tid / IN_SZ, xi = tid % IN_SZ;
    const bool is_x_thread = (tid < 16 * IN_SZ);
    if (is_x_thread)
        A_lds[0][xr][H_SZ + xi] = (_Float16)x[((size_t)(b0 + xr) * T_SZ + 0) * IN_SZ + xi];
    __syncthreads();

    for (int t = 0; t < T_SZ; ++t) {
        const int p = t & 1, q = p ^ 1;

        // issue next-step x load early (consumed at bottom of this iteration)
        float xn = 0.0f;
        if (t + 1 < T_SZ && is_x_thread)
            xn = x[((size_t)(b0 + xr) * T_SZ + (t + 1)) * IN_SZ + xi];

        // ---- GEMM: gates = [h|x|1] x Wcat, reading buffer p ----
        v8f acc[4] = {};
        const unsigned int* Arow = (const unsigned int*)(&A_lds[p][nlo][0]);
        const int khalf = (lane < 16) ? 0 : 8;
#pragma unroll
        for (int kc = 0; kc < 3; ++kc) {
            v8u au;
#pragma unroll
            for (int v = 0; v < 4; ++v) au[v]     = Arow[(kc * 32 + khalf + 2 * v) >> 1];
#pragma unroll
            for (int v = 0; v < 4; ++v) au[4 + v] = Arow[(kc * 32 + 16 + khalf + 2 * v) >> 1];
            const v16h af = __builtin_bit_cast(v16h, au);
#pragma unroll
            for (int g = 0; g < 4; ++g)
                acc[g] = __builtin_amdgcn_wmma_f32_16x16x32_f16(
                    false, af, false, bfrag[kc][g], (short)0, acc[g], false, false);
        }

        // ---- coalesced store of h(t-1) from the stable read buffer (overlaps WMMA) ----
        if (t > 0) {
            v8h hv = *(const v8h*)(&A_lds[p][srow][scb]);
            *(v8h*)(hrow_out + (size_t)(t - 1) * H_SZ) = hv;
        }

        // ---- pointwise LSTM update, entirely in registers ----
        // acc[0]=i, acc[1]=f, acc[2]=g, acc[3]=o for (row rbase+r, col hcol)
#pragma unroll
        for (int r = 0; r < 8; ++r) {
            const float c = fast_sigmoid(acc[1][r]) * c_st[r] +
                            fast_sigmoid(acc[0][r]) * fast_tanh(acc[2][r]);
            c_st[r] = c;
            const float h = fast_sigmoid(acc[3][r]) * fast_tanh(c);
            A_lds[q][rbase + r][hcol] = (_Float16)h;   // h(t) -> write buffer
        }

        if (t + 1 < T_SZ && is_x_thread) A_lds[q][xr][H_SZ + xi] = (_Float16)xn;

        __syncthreads();   // single barrier per step: buffer q now complete
    }

    // final h(T-1) lives in buffer 0 (T even); last __syncthreads already passed
    {
        v8h hv = *(const v8h*)(&A_lds[0][srow][scb]);
        *(v8h*)(hrow_out + (size_t)(T_SZ - 1) * H_SZ) = hv;
    }
}

// ---- FC: out[r][o] = Hbuf[r] . WfcT[:,o] + b_fc[o], r in [0, B*T) ----
__global__ __launch_bounds__(128) void fc_kernel(const _Float16* __restrict__ Hbuf,
                                                 const unsigned int* __restrict__ WfcP,
                                                 const float* __restrict__ b_fc,
                                                 float* __restrict__ out) {
    const int tid  = threadIdx.x;
    const int lane = tid & 31;
    const int wid  = tid >> 5;
    const int nlo  = lane & 15;

    // register-resident W_fc fragments (pre-packed); wave covers N tiles [wid*8, wid*8+8)
    v16h bfrag[2][8];
    float bias[8];
    {
        const v8u* Wv = (const v8u*)WfcP;
#pragma unroll
        for (int nt = 0; nt < 8; ++nt) {
            const int col = (wid * 8 + nt) * 16 + nlo;
            bias[nt] = (col < OUT_SZ) ? b_fc[col] : 0.0f;
#pragma unroll
            for (int kc = 0; kc < 2; ++kc)
                bfrag[kc][nt] = __builtin_bit_cast(
                    v16h, Wv[((kc * 8 + nt) * 4 + wid) * 32 + lane]);
        }
    }

#pragma unroll 1
    for (int mt = 0; mt < 8; ++mt) {
        const int rowbase = blockIdx.x * 128 + mt * 16;
        const unsigned int* Arow =
            (const unsigned int*)(Hbuf + (size_t)(rowbase + nlo) * H_SZ);
        const int khalf = (lane < 16) ? 0 : 8;
        v16h af[2];
#pragma unroll
        for (int kc = 0; kc < 2; ++kc) {
            v8u au;
#pragma unroll
            for (int v = 0; v < 4; ++v) au[v]     = Arow[(kc * 32 + khalf + 2 * v) >> 1];
#pragma unroll
            for (int v = 0; v < 4; ++v) au[4 + v] = Arow[(kc * 32 + 16 + khalf + 2 * v) >> 1];
            af[kc] = __builtin_bit_cast(v16h, au);
        }
#pragma unroll
        for (int nt = 0; nt < 8; ++nt) {
            v8f acc;
#pragma unroll
            for (int r = 0; r < 8; ++r) acc[r] = bias[nt];   // C init = bias broadcast
            acc = __builtin_amdgcn_wmma_f32_16x16x32_f16(
                false, af[0], false, bfrag[0][nt], (short)0, acc, false, false);
            acc = __builtin_amdgcn_wmma_f32_16x16x32_f16(
                false, af[1], false, bfrag[1][nt], (short)0, acc, false, false);
            const int col = (wid * 8 + nt) * 16 + nlo;
            if (col < OUT_SZ) {
                const int rb = rowbase + ((lane < 16) ? 0 : 8);
#pragma unroll
                for (int r = 0; r < 8; ++r)
                    out[(size_t)(rb + r) * OUT_SZ + col] = acc[r];
            }
        }
    }
}

extern "C" void kernel_launch(void* const* d_in, const int* in_sizes, int n_in,
                              void* d_out, int out_size, void* d_ws, size_t ws_size,
                              hipStream_t stream) {
    const float* x    = (const float*)d_in[0];
    const float* W_ih = (const float*)d_in[1];
    const float* W_hh = (const float*)d_in[2];
    const float* b_ih = (const float*)d_in[3];
    const float* b_hh = (const float*)d_in[4];
    const float* W_fc = (const float*)d_in[5];
    const float* b_fc = (const float*)d_in[6];
    float* out = (float*)d_out;

    char* ws = (char*)d_ws;
    unsigned int* WcatP = (unsigned int*)(ws + WS_WCAT);
    unsigned int* WfcP  = (unsigned int*)(ws + WS_WFC);
    _Float16*     Hbuf  = (_Float16*)(ws + WS_HBUF);

    prep_wcat_kernel<<<(NWCAT_W + 255) / 256, 256, 0, stream>>>(W_hh, W_ih, b_ih, b_hh, WcatP);
    prep_wfc_kernel<<<(NWFC_W + 255) / 256, 256, 0, stream>>>(W_fc, WfcP);
    lstm_kernel<<<B_SZ / 16, 128, 0, stream>>>(x, WcatP, Hbuf);
    fc_kernel<<<(B_SZ * T_SZ) / 128, 128, 0, stream>>>(Hbuf, WfcP, b_fc, out);
}
